// ScaledDotProductAttention_50414326120854
// MI455X (gfx1250) — matrix operand order repaired and checked by reading, not compiled, not run on hardware
//
#include <hip/hip_runtime.h>
#include <hip/hip_bf16.h>
#include <math.h>

// CDNA5 (gfx1250) scaled-dot-product attention.
// wave32 + v_wmma_f32_16x16x32_f16 + TDM (tensor_load_to_lds) for V staging.
// Memory-bound: ~1.14 GB mandatory traffic (attn write + mask read dominate)
// => ~49us floor at 23.3 TB/s; 34.4 GFLOP of GEMM is far below WMMA ceilings,
// so f16 WMMA with f32 accumulation is the right precision point, and data
// movement (coalesced mask/attn streams, TDM-fed LDS) is what matters.

typedef __attribute__((ext_vector_type(16))) _Float16 v16h;
typedef __attribute__((ext_vector_type(8)))  float    v8f;
typedef __attribute__((ext_vector_type(4)))  unsigned int u32x4;
typedef __attribute__((ext_vector_type(8)))  int          i32x8;
typedef __attribute__((ext_vector_type(4)))  int          i32x4;

constexpr int Bc = 32;
constexpr int Lc = 2048;   // Lq == Lk
constexpr int Dc = 64;     // head dim
constexpr int S_STRIDE = 2049;  // float stride: 2049%64==1 -> column reads hit distinct LDS banks
constexpr int VF_STRIDE = 68;   // float stride for V tile (64 data + 4 pad dwords per row)

__global__ __launch_bounds__(256)
void sdpa_gfx1250_kernel(const float* __restrict__ q,
                         const float* __restrict__ k,
                         const float* __restrict__ v,
                         const int*   __restrict__ mask,
                         float* __restrict__ outp,
                         float* __restrict__ attnp)
{
    __shared__ _Float16 Qh[16 * Dc];                 //   2 KB
    __shared__ float    S [16 * S_STRIDE];           // 128 KB  scores -> probs
    __shared__ alignas(16) float Vf[64 * VF_STRIDE]; //  17 KB  V staging (f32, TDM-padded rows)
    __shared__ float    Oscr[4 * 256];               //   4 KB  cross-wave partials

    const int tid  = threadIdx.x;
    const int lane = tid & 31;     // wave32
    const int wave = tid >> 5;     // 8 waves
    const int m    = lane & 15;    // M (or N) index within 16
    const int hi   = lane >> 4;    // half-wave select

    const int b  = blockIdx.y;
    const int q0 = blockIdx.x * 16;

    const float* qp = q + ((size_t)b * Lc + q0) * Dc;
    const float* kp = k + (size_t)b * Lc * Dc;
    const float* vp = v + (size_t)b * Lc * Dc;
    const int*   mp = mask + ((size_t)b * Lc + q0) * Lc;

    // ---------------- Phase 0: stage Q tile (16x64) as f16 ----------------
    for (int i = tid; i < 16 * Dc; i += 256)
        Qh[i] = (_Float16)qp[i];
    __syncthreads();

    // A fragments (16-bit A 16x32, this code's contiguous K order, same as B): lane m = M row,
    // element i -> K = c*32 + hi*16 + i
    v16h aq0, aq1;
    for (int i = 0; i < 16; ++i) {
        int kk = (hi << 4) + i;
        aq0[i] = Qh[m * Dc + kk];
        aq1[i] = Qh[m * Dc + 32 + kk];
    }

    // ---------------- Phase 1: S = mask(Q K^T / 8) ----------------
    // wave w covers key columns [w*256, w*256+256)
    for (int t = 0; t < 16; ++t) {
        const int n0 = wave * 256 + t * 16;
        // B fragment (32x16 f16): element i -> K = c*32 + hi*16 + i, N = m.
        // Per lane: 16 consecutive floats of K-row (n0+m) => b128 loads.
        const float* krow = kp + (size_t)(n0 + m) * Dc + hi * 16;
        if (t + 1 < 16)
            __builtin_prefetch(krow + 16 * Dc, 0, 0);   // global_prefetch_b8: next tile
        v16h bk0, bk1;
        for (int i = 0; i < 16; ++i) {
            bk0[i] = (_Float16)krow[i];
            bk1[i] = (_Float16)krow[32 + i];
        }
        v8f c = {};
        c = __builtin_amdgcn_wmma_f32_16x16x32_f16(false, aq0, false, bk0, (short)0, c, false, false);
        c = __builtin_amdgcn_wmma_f32_16x16x32_f16(false, aq1, false, bk1, (short)0, c, false, false);
        // C layout: element j -> row = hi*8 + j, col = n0 + m
        for (int j = 0; j < 8; ++j) {
            const int row = (hi << 3) + j;
            const int col = n0 + m;
            float s = c[j] * 0.125f;                       // / temperature (8.0)
            if (mp[(size_t)row * Lc + col] != 0) s = -INFINITY;
            S[row * S_STRIDE + col] = s;
        }
    }
    __syncthreads();

    // ---------------- Phase 2: row softmax (16 rows, 2 per wave) ----------------
    for (int rr = 0; rr < 2; ++rr) {
        const int r = wave * 2 + rr;
        float* Srow = S + r * S_STRIDE;
        float* Arow = attnp + ((size_t)b * Lc + q0 + r) * Lc;
        float mx = -INFINITY;
        for (int t = lane; t < Lc; t += 32) mx = fmaxf(mx, Srow[t]);
        for (int off = 16; off; off >>= 1) mx = fmaxf(mx, __shfl_xor(mx, off, 32));
        if (mx == -INFINITY) {
            // fully-masked row: softmax->NaN->1e-5->re-masked to 0 => all zeros
            for (int t = lane; t < Lc; t += 32) { Srow[t] = 0.f; Arow[t] = 0.f; }
        } else {
            float sum = 0.f;
            for (int t = lane; t < Lc; t += 32) {
                float e = __expf(Srow[t] - mx);   // masked entries: exp(-inf)=0
                Srow[t] = e; sum += e;
            }
            for (int off = 16; off; off >>= 1) sum += __shfl_xor(sum, off, 32);
            const float inv = 1.f / sum;
            for (int t = lane; t < Lc; t += 32) {
                float p = Srow[t] * inv;
                Srow[t] = p;
                Arow[t] = p;                      // coalesced attn stream-out
            }
        }
    }

    // ---------------- Phase 3: out = P @ V ----------------
    // 8 waves = 4 D-tiles (16 cols) x 2 K-halves; V staged 64 rows per step.
    // TDM (Tensor Data Mover) DMAs the 64x64 f32 tile global->LDS with a
    // 4-dword pad every 64 dwords (hardware-generated VF_STRIDE=68 rows).
    const int d0t = (wave & 3) * 16;
    const int kh  = wave >> 2;
    v8f acc = {};
#if __has_builtin(__builtin_amdgcn_tensor_load_to_lds)
    const unsigned ldsVf =
        (unsigned)(unsigned long long)(__attribute__((address_space(3))) float*)Vf;
#endif
    for (int k0 = 0; k0 < Lc; k0 += 64) {
        __syncthreads();                           // protect Vf reuse
#if __has_builtin(__builtin_amdgcn_tensor_load_to_lds)
        if (wave == 0) {
            const unsigned long long ga =
                (unsigned long long)(const void*)(vp + (size_t)k0 * Dc);
            u32x4 g0;
            g0[0] = 1u;                                        // count=1 (valid user D#)
            g0[1] = ldsVf;                                     // lds_addr
            g0[2] = (unsigned)(ga & 0xFFFFFFFFu);              // global_addr[31:0]
            g0[3] = (unsigned)((ga >> 32) & 0x01FFFFFFu)       // global_addr[56:32]
                  | (2u << 30);                                // type = 2 ("image")
            i32x8 g1;
            g1[0] = (2 << 16)                                  // data_size = 4 B
                  | (1 << 20)                                  // pad_enable
                  | (5 << 22)                                  // pad_interval: 64 dwords
                  | (3 << 25);                                 // pad_amount: 4 dwords
            g1[1] = (Dc & 0xFFFF) << 16;                       // tensor_dim0 = 64
            g1[2] = (64 & 0xFFFF) << 16;                       // tensor_dim1 = 64
            g1[3] = (Dc & 0xFFFF) << 16;                       // tile_dim0 = 64
            g1[4] = 64;                                        // tile_dim1 = 64, tile_dim2 = 0
            g1[5] = Dc;                                        // tensor_dim0_stride = 64
            g1[6] = 0;
            g1[7] = 0;
            i32x4 g2 = {0, 0, 0, 0};                           // dims 2/3 unused (2D tile)
            i32x4 g3 = {0, 0, 0, 0};
            i32x8 g4 = {0, 0, 0, 0, 0, 0, 0, 0};               // extra group (6-arg form)
            __builtin_amdgcn_tensor_load_to_lds(g0, g1, g2, g3, g4, 0);
            __builtin_amdgcn_s_wait_tensorcnt(0);              // s_wait_tensorcnt
        }
        __syncthreads();
#else
        for (int i = tid; i < 64 * Dc; i += 256) {
            int r = i >> 6, cc = i & 63;                       // coalesced fallback copy
            Vf[r * VF_STRIDE + cc] = vp[(size_t)(k0 + r) * Dc + cc];
        }
        __syncthreads();
#endif
        {
            const int kc = kh * 32;                            // local K base of this chunk
            v16h af, bf;
            for (int i = 0; i < 16; ++i) {
                int kl = kc + (hi << 4) + i;       // A: K index
                af[i] = (_Float16)S[m * S_STRIDE + k0 + kl];
                bf[i] = (_Float16)Vf[(kc + (hi << 4) + i) * VF_STRIDE + d0t + m];
            }
            acc = __builtin_amdgcn_wmma_f32_16x16x32_f16(false, af, false, bf, (short)0, acc, false, false);
        }
    }
    __syncthreads();
    if (kh == 1)
        for (int j = 0; j < 8; ++j)
            Oscr[(wave & 3) * 256 + j * 32 + lane] = acc[j];
    __syncthreads();
    if (kh == 0)
        for (int j = 0; j < 8; ++j) {
            float o = acc[j] + Oscr[(wave & 3) * 256 + j * 32 + lane];
            outp[((size_t)b * Lc + q0 + (hi << 3) + j) * Dc + d0t + m] = o;
        }
}

extern "C" void kernel_launch(void* const* d_in, const int* in_sizes, int n_in,
                              void* d_out, int out_size, void* d_ws, size_t ws_size,
                              hipStream_t stream) {
    (void)in_sizes; (void)n_in; (void)out_size; (void)d_ws; (void)ws_size;
    const float* q = (const float*)d_in[0];
    const float* k = (const float*)d_in[1];
    const float* v = (const float*)d_in[2];
    const int*   mask = (const int*)d_in[3];
    float* outp  = (float*)d_out;                         // [B, Lq, D]
    float* attnp = outp + (size_t)Bc * Lc * Dc;           // [B, Lq, Lk]
    dim3 grid(Lc / 16, Bc);   // 128 x 32 = 4096 workgroups
    dim3 block(256);          // 8 wave32 waves
    sdpa_gfx1250_kernel<<<grid, block, 0, stream>>>(q, k, v, mask, outp, attnp);
}
